// TinyLlamaBlock_6124623364699
// MI455X (gfx1250) — matrix operand order repaired and checked by reading, not compiled, not run on hardware
//
#include <hip/hip_runtime.h>

#define DIM   2048
#define NH    16
#define HD    128
#define BB    2
#define SS    2048
#define ROWS  (BB * SS)     // 4096
#define FFN   (4 * DIM)     // 8192
#define ATTN_SCALE 0.08838834764831845f  // 1/sqrt(HD)

typedef __attribute__((ext_vector_type(16))) __bf16 v16bf;
typedef __attribute__((ext_vector_type(8)))  float  v8f;

#ifndef __has_builtin
#define __has_builtin(x) 0
#endif
#if __has_builtin(__builtin_amdgcn_tensor_load_to_lds)
#define USE_TDM 1
#else
#define USE_TDM 0
#endif

// ---------------------------------------------------------------------------
// WMMA helpers (CDNA5: V_WMMA_F32_16X16X32_BF16, wave32)
// ---------------------------------------------------------------------------
__device__ __forceinline__ v8f wmma_bf16(v16bf a, v16bf b, v8f c) {
  return __builtin_amdgcn_wmma_f32_16x16x32_bf16(
      /*neg_a=*/false, a, /*neg_b=*/false, b,
      /*c_mod=*/(short)0, c, /*reuse_a=*/false, /*reuse_b=*/false);
}

// A-matrix (16x32 bf16): lane&15 = row M, hi = lane>>4 selects K chunks
// {hi*8 .. hi*8+7} and {16+hi*8 .. 16+hi*8+7}  (per ISA 16-bit A layout)
__device__ __forceinline__ v16bf load_frag_a(const __bf16* rowptr, int k, int hi) {
  union { uint4 u[2]; v16bf v; } r;
  r.u[0] = *reinterpret_cast<const uint4*>(rowptr + k + hi * 8);
  r.u[1] = *reinterpret_cast<const uint4*>(rowptr + k + 16 + hi * 8);
  return r.v;
}
// B-matrix (32x16 bf16): lane&15 = col N, hi selects K chunks
// {hi*8 .. hi*8+7} and {16+hi*8 .. 16+hi*8+7} (same K order as A).
__device__ __forceinline__ v16bf load_frag_b(const __bf16* colptr, int k, int hi) {
  union { uint4 u[2]; v16bf v; } r;
  r.u[0] = *reinterpret_cast<const uint4*>(colptr + k + hi * 8);
  r.u[1] = *reinterpret_cast<const uint4*>(colptr + k + 16 + hi * 8);
  return r.v;
}

__device__ __forceinline__ void wait_tensor0() {
#if __has_builtin(__builtin_amdgcn_s_wait_tensorcnt)
  __builtin_amdgcn_s_wait_tensorcnt(0);
#else
  asm volatile("s_wait_tensorcnt 0x0" ::: "memory");
#endif
}

#if USE_TDM
typedef unsigned int u32x4 __attribute__((ext_vector_type(4)));
typedef int i32x4 __attribute__((ext_vector_type(4)));
typedef int i32x8 __attribute__((ext_vector_type(8)));

// Tensor Data Mover: DMA a 32x32 bf16 tile (row stride K) from global to LDS.
// D# built per CDNA5 ISA 8.3/8.4: group0 = {count,lds_addr,global_addr,type},
// group1 = {data_size, tensor_dim0/1, tile_dim0/1, tensor_dim0_stride}.
__device__ __forceinline__ void tdm_load_a_tile(unsigned lds_addr, const __bf16* tile,
                                                int M, int K) {
  unsigned long long ga = (unsigned long long)(size_t)tile;
  u32x4 g0 = {};
  g0.x = 1u;                                                 // count=1 (user D#)
  g0.y = lds_addr;                                           // lds_addr
  g0.z = (unsigned)(ga & 0xffffffffu);                       // global_addr[31:0]
  g0.w = (unsigned)((ga >> 32) & 0x01ffffffu) | (2u << 30);  // [56:32] | type=2
  i32x8 g1 = {};
  g1.s0 = (1 << 16);                                         // data_size=2B
  g1.s1 = (int)(((unsigned)K & 0xffffu) << 16);              // tensor_dim0 lo
  g1.s2 = (int)((((unsigned)K >> 16) & 0xffffu) |
                (((unsigned)M & 0xffffu) << 16));            // dim0 hi | dim1 lo
  g1.s3 = (int)((((unsigned)M >> 16) & 0xffffu) |
                (32u << 16));                                // dim1 hi | tile_dim0=32
  g1.s4 = 32;                                                // tile_dim1=32
  g1.s5 = K;                                                 // tensor_dim0_stride
  g1.s6 = 0;
  g1.s7 = 0;
  i32x4 g2 = {};
  i32x4 g3 = {};
#if __clang_major__ >= 23
  i32x8 g4 = {};
  __builtin_amdgcn_tensor_load_to_lds(g0, g1, g2, g3, g4, 0);
#else
  __builtin_amdgcn_tensor_load_to_lds(g0, g1, g2, g3, 0);
#endif
}
#endif

// Fallback staging: 32 lanes, one 32-element (64B) row each.
__device__ __forceinline__ void stage_tile_global(const __bf16* gsrc, __bf16* dst,
                                                  int K, int lane) {
  const uint4* src = reinterpret_cast<const uint4*>(gsrc + (size_t)lane * K);
  uint4* d = reinterpret_cast<uint4*>(dst + lane * 32);
  d[0] = src[0]; d[1] = src[1]; d[2] = src[2]; d[3] = src[3];
}

// ---------------------------------------------------------------------------
// Weight convert + transpose: W[K][N] f32 -> Wt[N][K] bf16
// (bf16 weight set ~128 MB => resident in the 192 MB L2 across the GEMMs)
// ---------------------------------------------------------------------------
__global__ void wconv_kernel(const float* __restrict__ W, __bf16* __restrict__ Wt,
                             int K, int N) {
  size_t idx = (size_t)blockIdx.x * blockDim.x + threadIdx.x;
  if (idx >= (size_t)K * N) return;
  int k = (int)(idx / N);
  int n = (int)(idx % N);
  Wt[(size_t)n * K + k] = (__bf16)W[idx];
}

// ---------------------------------------------------------------------------
// RMSNorm: per-row, f32 in -> bf16 out
// ---------------------------------------------------------------------------
__global__ void rmsnorm_kernel(const float* __restrict__ x, const float* __restrict__ w,
                               __bf16* __restrict__ out) {
  int row = blockIdx.x;
  const float* xr = x + (size_t)row * DIM;
  __shared__ float red[256];
  float s = 0.f;
  for (int i = threadIdx.x; i < DIM; i += 256) { float v = xr[i]; s += v * v; }
  red[threadIdx.x] = s;
  __syncthreads();
  for (int o = 128; o > 0; o >>= 1) {
    if (threadIdx.x < (unsigned)o) red[threadIdx.x] += red[threadIdx.x + o];
    __syncthreads();
  }
  float inv = rsqrtf(red[0] / (float)DIM + 1e-6f);
  for (int i = threadIdx.x; i < DIM; i += 256)
    out[(size_t)row * DIM + i] = (__bf16)(xr[i] * inv * w[i]);
}

// ---------------------------------------------------------------------------
// GEMM: C[M][N] = A[M][K] (bf16) @ Wt[N][K] (bf16), f32 accumulate.
// Block = 4 waves, block tile 32M x 256N. The 32xK A panel is staged into a
// double-buffered LDS tile by the Tensor Data Mover (wave 0 issues the DMA,
// s_wait_tensorcnt + barrier publishes it); each wave computes a 32x64 tile
// (8 x v8f accumulators), B streamed from L2-resident bf16 weights.
// MODE: 0 = f32 out, 1 = bf16 out, 2 = f32 out + residual add
// ---------------------------------------------------------------------------
template <int MODE>
__global__ void __launch_bounds__(128, 2)
gemm_kernel(const __bf16* __restrict__ A, const __bf16* __restrict__ Bt,
            const float* __restrict__ res, float* __restrict__ Cf,
            __bf16* __restrict__ Cb, int M, int N, int K) {
  const int lane = threadIdx.x & 31;
  const int wave = threadIdx.x >> 5;
  const int tilesN = N >> 8;
  const int tm = blockIdx.x / tilesN;
  const int tn = blockIdx.x % tilesN;
  const int r = lane & 15;
  const int hi = lane >> 4;

  __shared__ __align__(16) __bf16 atile[2][32 * 32];

  const __bf16* Ablk = A + (size_t)tm * 32 * K;
  const int n0 = tn * 256 + wave * 64;
  const __bf16* brow = Bt + (size_t)(n0 + r) * K;

  if (wave == 0) {
#if USE_TDM
    tdm_load_a_tile((unsigned)(size_t)&atile[0][0], Ablk, M, K);
    wait_tensor0();
#else
    stage_tile_global(Ablk, &atile[0][0], K, lane);
#endif
  }
  __syncthreads();

  v8f acc[2][4] = {};
  for (int k = 0; k < K; k += 32) {
    const int cur = (k >> 5) & 1;
    if (wave == 0 && (k + 32) < K) {
#if USE_TDM
      tdm_load_a_tile((unsigned)(size_t)&atile[cur ^ 1][0], Ablk + (k + 32), M, K);
#else
      stage_tile_global(Ablk + (k + 32), &atile[cur ^ 1][0], K, lane);
#endif
    }
    const __bf16* a0 = &atile[cur][r * 32];
    v16bf af0 = load_frag_a(a0, 0, hi);
    v16bf af1 = load_frag_a(a0 + 16 * 32, 0, hi);
    __builtin_prefetch(brow + k + 512, 0, 0);  // global_prefetch_b8
#pragma unroll
    for (int j = 0; j < 4; ++j) {
      v16bf b = load_frag_b(brow + (size_t)j * 16 * K, k, hi);
      acc[0][j] = wmma_bf16(af0, b, acc[0][j]);
      acc[1][j] = wmma_bf16(af1, b, acc[1][j]);
    }
#if USE_TDM
    if (wave == 0) wait_tensor0();
#endif
    __syncthreads();
  }

#pragma unroll
  for (int mi = 0; mi < 2; ++mi) {
#pragma unroll
    for (int j = 0; j < 4; ++j) {
#pragma unroll
      for (int rr = 0; rr < 8; ++rr) {
        size_t row = (size_t)tm * 32 + mi * 16 + hi * 8 + rr;
        size_t col = (size_t)n0 + j * 16 + r;
        size_t idx = row * N + col;
        float v = acc[mi][j][rr];
        if (MODE == 2) v += res[idx];
        if (MODE == 1) Cb[idx] = (__bf16)v;
        else           Cf[idx] = v;
      }
    }
  }
}

// ---------------------------------------------------------------------------
// RoPE on q,k (f32 -> bf16) + transpose V into [b][h][d][s] bf16.
// Q is pre-scaled by 1/sqrt(HD) here so the attention inner loop has no
// per-element scale multiplies.
// ---------------------------------------------------------------------------
__global__ void rope_kernel(const float* __restrict__ q, const float* __restrict__ k,
                            const float* __restrict__ v, const float* __restrict__ cosb,
                            const float* __restrict__ sinb, __bf16* __restrict__ qb,
                            __bf16* __restrict__ kb, __bf16* __restrict__ vt) {
  size_t idx = (size_t)blockIdx.x * blockDim.x + threadIdx.x;
  if (idx >= (size_t)ROWS * DIM) return;
  int c = (int)(idx & (DIM - 1));
  int bs = (int)(idx >> 11);      // DIM = 2^11
  int s = bs & (SS - 1);
  int b = bs >> 11;               // SS = 2^11
  int h = c >> 7;                 // HD = 128
  int d = c & (HD - 1);
  float cs = cosb[s * HD + d];
  float sn = sinb[s * HD + d];
  float qv = q[idx], kv = k[idx];
  float qr, kr;
  if (d < HD / 2) { qr = -q[idx + HD / 2]; kr = -k[idx + HD / 2]; }
  else            { qr =  q[idx - HD / 2]; kr =  k[idx - HD / 2]; }
  qb[idx] = (__bf16)((qv * cs + qr * sn) * ATTN_SCALE);
  kb[idx] = (__bf16)(kv * cs + kr * sn);
  vt[((size_t)(b * NH + h) * HD + d) * SS + s] = (__bf16)v[idx];
}

// ---------------------------------------------------------------------------
// One 32-key attention block. MASKED=false instantiation statically drops
// the causal compare/select chain (used for all fully-unmasked blocks).
// ---------------------------------------------------------------------------
template <bool MASKED>
__device__ __forceinline__ void attn_block(
    int kb, int qbase, const v16bf (&qf)[4], v8f (&o)[8],
    float (&mrow)[8], float (&lrow)[8],
    const __bf16* Kbase, const __bf16* Vbase, __bf16* pw,
    int r, int hi) {
  v8f s0 = {}, s1 = {};
  const __bf16* krow0 = Kbase + (size_t)(kb + r) * DIM;
  const __bf16* krow1 = krow0 + (size_t)16 * DIM;
#pragma unroll
  for (int kc = 0; kc < 4; ++kc) {
    s0 = wmma_bf16(qf[kc], load_frag_b(krow0, kc * 32, hi), s0);
    s1 = wmma_bf16(qf[kc], load_frag_b(krow1, kc * 32, hi), s1);
  }

#pragma unroll
  for (int rr = 0; rr < 8; ++rr) {
    float v0 = s0[rr];
    float v1 = s1[rr];
    if (MASKED) {
      int qi = qbase + rr;
      v0 = (kb + r      <= qi) ? v0 : -1e30f;
      v1 = (kb + 16 + r <= qi) ? v1 : -1e30f;
    }
    float mx = fmaxf(v0, v1);
#pragma unroll
    for (int off = 8; off; off >>= 1) mx = fmaxf(mx, __shfl_xor(mx, off, 16));
    float mnew = fmaxf(mrow[rr], mx);
    float alpha = __expf(mrow[rr] - mnew);
    mrow[rr] = mnew;
    float p0 = __expf(v0 - mnew);
    float p1 = __expf(v1 - mnew);
    float sum = p0 + p1;
#pragma unroll
    for (int off = 8; off; off >>= 1) sum += __shfl_xor(sum, off, 16);
    lrow[rr] = lrow[rr] * alpha + sum;
#pragma unroll
    for (int j = 0; j < 8; ++j) o[j][rr] *= alpha;
    int m = hi * 8 + rr;
    pw[m * 32 + r]      = (__bf16)p0;
    pw[m * 32 + 16 + r] = (__bf16)p1;
  }
  asm volatile("s_wait_dscnt 0x0" ::: "memory");

  v16bf pa = load_frag_a(pw + r * 32, 0, hi);
#pragma unroll
  for (int j = 0; j < 8; ++j) {
    const __bf16* vrow = Vbase + (size_t)(j * 16 + r) * SS;
    o[j] = wmma_bf16(pa, load_frag_b(vrow, kb, hi), o[j]);
  }
}

// ---------------------------------------------------------------------------
// Flash attention: one wave per 16-row q tile, streaming 32-key blocks.
// Full blocks run maskless; exactly one diagonal block per q-tile pays the
// causal compare/select. Online softmax lives entirely in the WMMA C/D
// register layout (rows in 16-lane halves -> shfl_xor width-16 reductions);
// P reshaped via per-wave LDS (s_wait_dscnt). __launch_bounds__(128,1)
// guarantees the full VGPR budget (no scratch spills).
// ---------------------------------------------------------------------------
__global__ void __launch_bounds__(128, 1)
attn_kernel(const __bf16* __restrict__ Q, const __bf16* __restrict__ Kc,
            const __bf16* __restrict__ Vt, __bf16* __restrict__ O) {
  const int lane = threadIdx.x & 31;
  const int wave = threadIdx.x >> 5;
  const int qt = blockIdx.x * 4 + wave;  // q tile (16 rows)
  const int h = blockIdx.y;
  const int b = blockIdx.z;
  const int r = lane & 15;
  const int hi = lane >> 4;

  __shared__ __align__(16) __bf16 pstage[4][16 * 32];
  __bf16* pw = pstage[wave];

  const __bf16* Qrow = Q + (size_t)(b * SS + qt * 16 + r) * DIM + h * HD;
  v16bf qf[4];
#pragma unroll
  for (int kc = 0; kc < 4; ++kc) qf[kc] = load_frag_a(Qrow, kc * 32, hi);

  v8f o[8] = {};
  float mrow[8], lrow[8];
#pragma unroll
  for (int rr = 0; rr < 8; ++rr) { mrow[rr] = -1e30f; lrow[rr] = 0.f; }

  const int qbase = qt * 16 + hi * 8;
  const __bf16* Kbase = Kc + (size_t)(b * SS) * DIM + h * HD;
  const __bf16* Vbase = Vt + (size_t)(b * NH + h) * HD * SS;

  const int kfull = (qt * 16) & ~31;     // keys [0, kfull) need no mask
  for (int kb = 0; kb < kfull; kb += 32)
    attn_block<false>(kb, qbase, qf, o, mrow, lrow, Kbase, Vbase, pw, r, hi);
  attn_block<true>(kfull, qbase, qf, o, mrow, lrow, Kbase, Vbase, pw, r, hi);

  __bf16* Orow = O + (size_t)(b * SS + qt * 16 + hi * 8) * DIM + h * HD;
#pragma unroll
  for (int rr = 0; rr < 8; ++rr) {
    float invl = 1.f / lrow[rr];
#pragma unroll
    for (int j = 0; j < 8; ++j)
      Orow[(size_t)rr * DIM + j * 16 + r] = (__bf16)(o[j][rr] * invl);
  }
}

// ---------------------------------------------------------------------------
// SwiGLU: silu(g1) * g3, bf16 in/out
// ---------------------------------------------------------------------------
__global__ void swiglu_kernel(const __bf16* __restrict__ g1, const __bf16* __restrict__ g3,
                              __bf16* __restrict__ out, size_t total) {
  size_t idx = (size_t)blockIdx.x * blockDim.x + threadIdx.x;
  if (idx >= total) return;
  float a = (float)g1[idx];
  float c = (float)g3[idx];
  out[idx] = (__bf16)((a / (1.f + __expf(-a))) * c);
}

// ---------------------------------------------------------------------------
// Host orchestration
// ---------------------------------------------------------------------------
extern "C" void kernel_launch(void* const* d_in, const int* in_sizes, int n_in,
                              void* d_out, int out_size, void* d_ws, size_t ws_size,
                              hipStream_t stream) {
  (void)in_sizes; (void)n_in; (void)out_size; (void)ws_size;
  const float* x      = (const float*)d_in[0];
  const float* cosb   = (const float*)d_in[1];
  const float* sinb   = (const float*)d_in[2];
  /* d_in[3] = additive causal mask; handled analytically in attn_kernel */
  const float* w_attn = (const float*)d_in[4];
  const float* w_ffn  = (const float*)d_in[5];
  const float* wq     = (const float*)d_in[6];
  const float* wk     = (const float*)d_in[7];
  const float* wv     = (const float*)d_in[8];
  const float* wo     = (const float*)d_in[9];
  const float* w1     = (const float*)d_in[10];
  const float* w2     = (const float*)d_in[11];
  const float* w3     = (const float*)d_in[12];
  float* out = (float*)d_out;

  char* p = (char*)d_ws;
  auto alloc = [&](size_t bytes) -> void* {
    void* q = (void*)p;
    p += (bytes + 255) & ~(size_t)255;
    return q;
  };
  __bf16* wqt = (__bf16*)alloc((size_t)DIM * DIM * 2);
  __bf16* wkt = (__bf16*)alloc((size_t)DIM * DIM * 2);
  __bf16* wvt = (__bf16*)alloc((size_t)DIM * DIM * 2);
  __bf16* wot = (__bf16*)alloc((size_t)DIM * DIM * 2);
  __bf16* w1t = (__bf16*)alloc((size_t)DIM * FFN * 2);
  __bf16* w3t = (__bf16*)alloc((size_t)DIM * FFN * 2);
  __bf16* w2t = (__bf16*)alloc((size_t)FFN * DIM * 2);
  __bf16* hb  = (__bf16*)alloc((size_t)ROWS * DIM * 2);
  float*  qf  = (float*)alloc((size_t)ROWS * DIM * 4);
  float*  kf  = (float*)alloc((size_t)ROWS * DIM * 4);
  float*  vf  = (float*)alloc((size_t)ROWS * DIM * 4);
  __bf16* qb  = (__bf16*)alloc((size_t)ROWS * DIM * 2);
  __bf16* kb  = (__bf16*)alloc((size_t)ROWS * DIM * 2);
  __bf16* vt  = (__bf16*)alloc((size_t)ROWS * DIM * 2);
  __bf16* ob  = (__bf16*)alloc((size_t)ROWS * DIM * 2);
  float*  x1  = (float*)alloc((size_t)ROWS * DIM * 4);
  __bf16* h2  = (__bf16*)alloc((size_t)ROWS * DIM * 2);
  __bf16* g1  = (__bf16*)alloc((size_t)ROWS * FFN * 2);
  __bf16* g3  = (__bf16*)alloc((size_t)ROWS * FFN * 2);
  __bf16* ff  = (__bf16*)alloc((size_t)ROWS * FFN * 2);

  auto eblocks = [](size_t n) { return (unsigned)((n + 255) / 256); };
  auto gblocks = [](int M, int N) { return (unsigned)((M / 32) * (N / 256)); };

  wconv_kernel<<<eblocks((size_t)DIM * DIM), 256, 0, stream>>>(wq, wqt, DIM, DIM);
  wconv_kernel<<<eblocks((size_t)DIM * DIM), 256, 0, stream>>>(wk, wkt, DIM, DIM);
  wconv_kernel<<<eblocks((size_t)DIM * DIM), 256, 0, stream>>>(wv, wvt, DIM, DIM);
  wconv_kernel<<<eblocks((size_t)DIM * DIM), 256, 0, stream>>>(wo, wot, DIM, DIM);
  wconv_kernel<<<eblocks((size_t)DIM * FFN), 256, 0, stream>>>(w1, w1t, DIM, FFN);
  wconv_kernel<<<eblocks((size_t)DIM * FFN), 256, 0, stream>>>(w3, w3t, DIM, FFN);
  wconv_kernel<<<eblocks((size_t)FFN * DIM), 256, 0, stream>>>(w2, w2t, FFN, DIM);

  // Attention path
  rmsnorm_kernel<<<ROWS, 256, 0, stream>>>(x, w_attn, hb);
  gemm_kernel<0><<<gblocks(ROWS, DIM), 128, 0, stream>>>(hb, wqt, nullptr, qf, nullptr, ROWS, DIM, DIM);
  gemm_kernel<0><<<gblocks(ROWS, DIM), 128, 0, stream>>>(hb, wkt, nullptr, kf, nullptr, ROWS, DIM, DIM);
  gemm_kernel<0><<<gblocks(ROWS, DIM), 128, 0, stream>>>(hb, wvt, nullptr, vf, nullptr, ROWS, DIM, DIM);
  rope_kernel<<<eblocks((size_t)ROWS * DIM), 256, 0, stream>>>(qf, kf, vf, cosb, sinb, qb, kb, vt);
  attn_kernel<<<dim3(SS / 16 / 4, NH, BB), 128, 0, stream>>>(qb, kb, vt, ob);
  gemm_kernel<2><<<gblocks(ROWS, DIM), 128, 0, stream>>>(ob, wot, x, x1, nullptr, ROWS, DIM, DIM);

  // FFN path
  rmsnorm_kernel<<<ROWS, 256, 0, stream>>>(x1, w_ffn, h2);
  gemm_kernel<1><<<gblocks(ROWS, FFN), 128, 0, stream>>>(h2, w1t, nullptr, nullptr, g1, ROWS, FFN, DIM);
  gemm_kernel<1><<<gblocks(ROWS, FFN), 128, 0, stream>>>(h2, w3t, nullptr, nullptr, g3, ROWS, FFN, DIM);
  swiglu_kernel<<<eblocks((size_t)ROWS * FFN), 256, 0, stream>>>(g1, g3, ff, (size_t)ROWS * FFN);
  gemm_kernel<2><<<gblocks(ROWS, DIM), 128, 0, stream>>>(ff, w2t, x1, out, nullptr, ROWS, DIM, FFN);
}
